// TransformerBlock_17583596109998
// MI455X (gfx1250) — matrix operand order repaired and checked by reading, not compiled, not run on hardware
//
#include <hip/hip_runtime.h>

// ---------------------------------------------------------------- constants
#define EMB   1024
#define HEADS 16
#define HDIM  64
#define FFDIM 4096
#define BATCH 4
#define SEQ   2048
#define ROWS  (BATCH * SEQ)   // 8192

typedef __attribute__((ext_vector_type(16))) __bf16 v16bf;
typedef __attribute__((ext_vector_type(8)))  float  v8f;

union Frag16 { uint4 q[2]; v16bf v; };   // 32 bytes = 8 VGPRs of packed bf16

__device__ __forceinline__ unsigned short f2bf(float f) {
  union { float f; unsigned int u; } c; c.f = f;
  unsigned int u = c.u;
  u += 0x7FFFu + ((u >> 16) & 1u);       // round-to-nearest-even
  return (unsigned short)(u >> 16);
}

// CDNA5 async global->LDS copy (ASYNCcnt-tracked, bypasses VGPRs).
// Copies 64 contiguous bytes (4 x b128) per calling lane.
__device__ __forceinline__ void async_copy64(const unsigned short* g, unsigned lds) {
#pragma unroll
  for (int c = 0; c < 4; ++c)
    asm volatile("global_load_async_to_lds_b128 %0, %1, off"
                 :: "v"(lds + c * 16), "v"(g + c * 8) : "memory");
}
__device__ __forceinline__ void wait_async0() {
  asm volatile("s_wait_asynccnt 0x0" ::: "memory");
}
__device__ __forceinline__ unsigned lds_addr(const void* p) {
  // flat LDS address: low 32 bits are the wave-relative LDS byte offset
  return (unsigned)(unsigned long long)(p);
}

// ------------------------------------------------ weight transpose f32->bf16
// in: K x N (row major)   out: N x K (row major) == B^T for WMMA B-fragments
__global__ void transpose_cast_bf16(const float* __restrict__ in,
                                    unsigned short* __restrict__ out,
                                    int K, int N) {
  __shared__ float tile[32][33];
  const int k0 = blockIdx.x * 32, n0 = blockIdx.y * 32;
  const int tx = threadIdx.x & 31, ty = threadIdx.x >> 5;   // 32 x 8
#pragma unroll
  for (int i = 0; i < 4; ++i)
    tile[ty + i * 8][tx] = in[(size_t)(k0 + ty + i * 8) * N + (n0 + tx)];
  __syncthreads();
#pragma unroll
  for (int i = 0; i < 4; ++i)
    out[(size_t)(n0 + ty + i * 8) * K + (k0 + tx)] = f2bf(tile[tx][ty + i * 8]);
}

// ------------------------------------------------------- layernorm -> bf16
__global__ void layernorm_bf16(const float* __restrict__ x,
                               const float* __restrict__ g,
                               const float* __restrict__ b,
                               unsigned short* __restrict__ out) {
  __shared__ float rsum[256], rsq[256];
  const int row = blockIdx.x, t = threadIdx.x;
  const float4 v4 = ((const float4*)(x + (size_t)row * EMB))[t];
  rsum[t] = v4.x + v4.y + v4.z + v4.w;
  rsq[t]  = v4.x * v4.x + v4.y * v4.y + v4.z * v4.z + v4.w * v4.w;
  __syncthreads();
  for (int off = 128; off > 0; off >>= 1) {
    if (t < off) { rsum[t] += rsum[t + off]; rsq[t] += rsq[t + off]; }
    __syncthreads();
  }
  const float mean = rsum[0] * (1.0f / EMB);
  const float var  = rsq[0] * (1.0f / EMB) - mean * mean;
  const float rstd = rsqrtf(var + 1e-5f);
  const float vv[4] = {v4.x, v4.y, v4.z, v4.w};
#pragma unroll
  for (int i = 0; i < 4; ++i) {
    const int c = t * 4 + i;
    out[(size_t)row * EMB + c] = f2bf((vv[i] - mean) * rstd * g[c] + b[c]);
  }
}

// --------------------------------------------------------------- WMMA GEMM
// C[M,N] = A[M,K](bf16) * BT[N,K]^T(bf16) + bias, with fused epilogues.
// 128x128 block tile, BK=64, double-buffered LDS fed by async-to-LDS copies.
enum { MODE_QK = 0, MODE_VT = 1, MODE_GELU = 2, MODE_F32RES = 3 };

template <int MODE>
__global__ __launch_bounds__(256)
void gemm_bf16_wmma(const unsigned short* __restrict__ A,
                    const unsigned short* __restrict__ BT,
                    const float* __restrict__ bias,
                    const float* __restrict__ resid,
                    void* __restrict__ outp,
                    int M, int N, int K) {
  constexpr int LDT = 72;                       // padded LDS row stride (ushorts)
  __shared__ __attribute__((aligned(16))) unsigned short ldsA[2][128 * LDT];
  __shared__ __attribute__((aligned(16))) unsigned short ldsB[2][128 * LDT];
  const int m0 = blockIdx.y * 128, n0 = blockIdx.x * 128;
  const int t = threadIdx.x;
  const int lane = t & 31;
  const int wave = t >> 5;                      // 8 waves
  const int wm = (wave & 3) * 32;               // wave tile: 32 (M) x 64 (N)
  const int wn = (wave >> 2) * 64;
  const int lm = lane & 15, half = lane >> 4;

  const v8f vzero = {};
  v8f acc[2][4];
#pragma unroll
  for (int i = 0; i < 2; ++i)
#pragma unroll
    for (int j = 0; j < 4; ++j) acc[i][j] = vzero;

  const int ldrow = t >> 1;                     // 0..127
  const int ldk   = (t & 1) * 32;               // 0 / 32 (ushorts)
  const unsigned short* gA0 = A  + (size_t)(m0 + ldrow) * K + ldk;
  const unsigned short* gB0 = BT + (size_t)(n0 + ldrow) * K + ldk;
  const unsigned laddrA[2] = {lds_addr(&ldsA[0][ldrow * LDT + ldk]),
                              lds_addr(&ldsA[1][ldrow * LDT + ldk])};
  const unsigned laddrB[2] = {lds_addr(&ldsB[0][ldrow * LDT + ldk]),
                              lds_addr(&ldsB[1][ldrow * LDT + ldk])};

  // prime buffer 0
  async_copy64(gA0, laddrA[0]);
  async_copy64(gB0, laddrB[0]);

  int buf = 0;
  for (int k0 = 0; k0 < K; k0 += 64, buf ^= 1) {
    wait_async0();          // own async writes done
    __syncthreads();        // everyone's async writes done; prev reads of buf^1 done
    if (k0 + 64 < K) {      // stream next tile into the other buffer
      async_copy64(gA0 + k0 + 64, laddrA[buf ^ 1]);
      async_copy64(gB0 + k0 + 64, laddrB[buf ^ 1]);
    }
#pragma unroll
    for (int ks = 0; ks < 64; ks += 32) {       // two WMMA K-steps per tile
      Frag16 fa[2], fb[4];
#pragma unroll
      for (int i = 0; i < 2; ++i) {             // A frag: k = half*16 + 0..15
        const unsigned short* p = &ldsA[buf][(wm + i * 16 + lm) * LDT + ks];
        fa[i].q[0] = *(const uint4*)(p + half * 16);
        fa[i].q[1] = *(const uint4*)(p + half * 16 + 8);
      }
#pragma unroll
      for (int j = 0; j < 4; ++j) {             // B frag: k = half*16 + 0..15
        const unsigned short* p = &ldsB[buf][(wn + j * 16 + lm) * LDT + ks + half * 16];
        fb[j].q[0] = *(const uint4*)(p);
        fb[j].q[1] = *(const uint4*)(p + 8);
      }
#pragma unroll
      for (int i = 0; i < 2; ++i)
#pragma unroll
        for (int j = 0; j < 4; ++j)
          acc[i][j] = __builtin_amdgcn_wmma_f32_16x16x32_bf16(
              false, fa[i].v, false, fb[j].v, (short)0, acc[i][j], false, false);
    }
  }

  // epilogue: VGPR r <-> row half*8+r, lane <-> col
#pragma unroll
  for (int i = 0; i < 2; ++i) {
#pragma unroll
    for (int j = 0; j < 4; ++j) {
#pragma unroll
      for (int r = 0; r < 8; ++r) {
        const int m = m0 + wm + i * 16 + half * 8 + r;
        const int n = n0 + wn + j * 16 + lm;
        float val = acc[i][j][r] + bias[n];
        if (MODE == MODE_QK) {                  // -> [b, h, s, d] bf16
          const int bi = m >> 11, si = m & (SEQ - 1);
          const int h = n >> 6, d = n & 63;
          ((unsigned short*)outp)[(((size_t)(bi * HEADS + h) * SEQ + si) << 6) + d] = f2bf(val);
        } else if (MODE == MODE_VT) {           // -> [b, h, d, s] bf16 (V^T)
          const int bi = m >> 11, si = m & (SEQ - 1);
          const int h = n >> 6, d = n & 63;
          ((unsigned short*)outp)[(((size_t)((bi * HEADS + h) * HDIM + d)) << 11) + si] = f2bf(val);
        } else if (MODE == MODE_GELU) {         // tanh-GELU -> bf16 row major
          const float u = 0.7978845608028654f * (val + 0.044715f * val * val * val);
          const float gl = 0.5f * val * (1.0f + tanhf(u));
          ((unsigned short*)outp)[(size_t)m * N + n] = f2bf(gl);
        } else {                                // f32 + residual
          ((float*)outp)[(size_t)m * N + n] = val + resid[(size_t)m * N + n];
        }
      }
    }
  }
}

// -------------------------------------------------- flash attention (causal)
// One wave per 16-query tile; key tiles processed in pairs (K=32 per WMMA).
__global__ __launch_bounds__(128)
void attn_flash_wmma(const unsigned short* __restrict__ Q,
                     const unsigned short* __restrict__ Kh,
                     const unsigned short* __restrict__ Vt,
                     unsigned short* __restrict__ ctx) {
  __shared__ unsigned short Pbuf[4 * 16 * 32];     // per-wave 16x32 P strip
  const int lane = threadIdx.x & 31;
  const int wave = threadIdx.x >> 5;
  const int lm = lane & 15, half = lane >> 4;
  const int bh = blockIdx.y;
  const int qt = blockIdx.x * 4 + wave;            // 16-query tile index
  unsigned short* P = &Pbuf[wave * 512];
  const v8f vzero = {};

  // Q fragments: 16 x 64, two K-chunks of 32
  Frag16 fq[2];
  const unsigned short* qrow = Q + (((size_t)bh * SEQ + qt * 16 + lm) << 6);
#pragma unroll
  for (int i = 0; i < 2; ++i) {
    fq[i].q[0] = *(const uint4*)(qrow + i * 32 + half * 16);
    fq[i].q[1] = *(const uint4*)(qrow + i * 32 + half * 16 + 8);
  }

  float mrun[8], lrun[8];
  v8f accO[4];
#pragma unroll
  for (int r = 0; r < 8; ++r) { mrun[r] = -3.0e38f; lrun[r] = 0.0f; }
#pragma unroll
  for (int j = 0; j < 4; ++j) accO[j] = vzero;

  const int qbase = qt * 16;
  for (int p = 0; p <= (qt >> 1); ++p) {
    float sc[2][8];
#pragma unroll
    for (int tile = 0; tile < 2; ++tile) {
      const int key = (p * 2 + tile) * 16 + lm;    // this lane's key column
      const unsigned short* krow = Kh + (((size_t)bh * SEQ + key) << 6);
      Frag16 fk0, fk1;                              // B frag over d (K dim)
      fk0.q[0] = *(const uint4*)(krow + half * 16);
      fk0.q[1] = *(const uint4*)(krow + half * 16 + 8);
      fk1.q[0] = *(const uint4*)(krow + 32 + half * 16);
      fk1.q[1] = *(const uint4*)(krow + 32 + half * 16 + 8);
      v8f s = __builtin_amdgcn_wmma_f32_16x16x32_bf16(
          false, fq[0].v, false, fk0.v, (short)0, vzero, false, false);
      s = __builtin_amdgcn_wmma_f32_16x16x32_bf16(
          false, fq[1].v, false, fk1.v, (short)0, s, false, false);
#pragma unroll
      for (int r = 0; r < 8; ++r) {
        const int qi = qbase + half * 8 + r;
        const float v = s[r] * 0.125f;             // 1/sqrt(64)
        sc[tile][r] = (key > qi) ? -3.0e38f : v;   // causal mask
      }
    }

    // online softmax: butterfly over the 16 lanes holding each row
    float corr[8];
#pragma unroll
    for (int r = 0; r < 8; ++r) {
      float rm = fmaxf(sc[0][r], sc[1][r]);
#pragma unroll
      for (int off = 1; off < 16; off <<= 1)
        rm = fmaxf(rm, __shfl_xor(rm, off, 32));
      const float mnew = fmaxf(mrun[r], rm);
      corr[r] = __expf(mrun[r] - mnew);
      const float p0 = __expf(sc[0][r] - mnew);
      const float p1 = __expf(sc[1][r] - mnew);
      float ps = p0 + p1;
#pragma unroll
      for (int off = 1; off < 16; off <<= 1)
        ps += __shfl_xor(ps, off, 32);
      lrun[r] = lrun[r] * corr[r] + ps;
      mrun[r] = mnew;
      P[(half * 8 + r) * 32 + lm]      = f2bf(p0); // C-layout -> row-major LDS
      P[(half * 8 + r) * 32 + 16 + lm] = f2bf(p1);
    }

    // P (16x32) as A-fragment from LDS; same-wave DS ops are in order.
    Frag16 fp;
    __builtin_memcpy(&fp.q[0], P + lm * 32 + half * 16, 16);
    __builtin_memcpy(&fp.q[1], P + lm * 32 + half * 16 + 8, 16);

#pragma unroll
    for (int j = 0; j < 4; ++j) {                  // d chunks of 16
      const int d = j * 16 + lm;
      const unsigned short* vrow =
          Vt + (((size_t)bh * HDIM + d) << 11) + p * 32 + half * 16;
      Frag16 fv;
      fv.q[0] = *(const uint4*)(vrow);
      fv.q[1] = *(const uint4*)(vrow + 8);
#pragma unroll
      for (int r = 0; r < 8; ++r) accO[j][r] *= corr[r];
      accO[j] = __builtin_amdgcn_wmma_f32_16x16x32_bf16(
          false, fp.v, false, fv.v, (short)0, accO[j], false, false);
    }
  }

  // normalize + store ctx as [b, s, h*64+d] bf16
  const int b = bh >> 4, h = bh & 15;
#pragma unroll
  for (int j = 0; j < 4; ++j) {
#pragma unroll
    for (int r = 0; r < 8; ++r) {
      const int qi = qbase + half * 8 + r;
      const size_t idx = (((size_t)(b * SEQ + qi)) << 10) + h * 64 + j * 16 + lm;
      ctx[idx] = f2bf(accO[j][r] / lrun[r]);
    }
  }
}

// ---------------------------------------------------------------- launcher
extern "C" void kernel_launch(void* const* d_in, const int* in_sizes, int n_in,
                              void* d_out, int out_size, void* d_ws, size_t ws_size,
                              hipStream_t stream) {
  (void)in_sizes; (void)n_in; (void)out_size; (void)ws_size;
  const float* x   = (const float*)d_in[0];
  const float* Wq  = (const float*)d_in[1];
  const float* bq  = (const float*)d_in[2];
  const float* Wk  = (const float*)d_in[3];
  const float* bk  = (const float*)d_in[4];
  const float* Wv  = (const float*)d_in[5];
  const float* bv  = (const float*)d_in[6];
  const float* Wo  = (const float*)d_in[7];
  const float* bo  = (const float*)d_in[8];
  const float* W1  = (const float*)d_in[9];
  const float* b1  = (const float*)d_in[10];
  const float* W2  = (const float*)d_in[11];
  const float* b2  = (const float*)d_in[12];
  const float* g1  = (const float*)d_in[13];
  const float* be1 = (const float*)d_in[14];
  const float* g2  = (const float*)d_in[15];
  const float* be2 = (const float*)d_in[16];
  float* out = (float*)d_out;

  char* ws = (char*)d_ws;
  const size_t MB = 1ull << 20;
  unsigned short* WqT = (unsigned short*)(ws + 0 * MB);    //  2 MB
  unsigned short* WkT = (unsigned short*)(ws + 2 * MB);    //  2 MB
  unsigned short* WvT = (unsigned short*)(ws + 4 * MB);    //  2 MB
  unsigned short* WoT = (unsigned short*)(ws + 6 * MB);    //  2 MB
  unsigned short* W1T = (unsigned short*)(ws + 8 * MB);    //  8 MB
  unsigned short* W2T = (unsigned short*)(ws + 16 * MB);   //  8 MB
  unsigned short* hbf = (unsigned short*)(ws + 24 * MB);   // 16 MB (LN1 out, reused for LN2 out)
  unsigned short* Qh  = (unsigned short*)(ws + 40 * MB);   // 16 MB
  unsigned short* Kh  = (unsigned short*)(ws + 56 * MB);   // 16 MB
  unsigned short* Vt  = (unsigned short*)(ws + 72 * MB);   // 16 MB
  unsigned short* ctx = (unsigned short*)(ws + 88 * MB);   // 16 MB
  unsigned short* a1  = (unsigned short*)(ws + 40 * MB);   // 64 MB, reuses dead Q/K/V/ctx
  float*          x1  = (float*)(ws + 104 * MB);           // 32 MB

  // 1) weight transposes (f32 -> bf16 W^T)
  transpose_cast_bf16<<<dim3(EMB / 32, EMB / 32), 256, 0, stream>>>(Wq, WqT, EMB, EMB);
  transpose_cast_bf16<<<dim3(EMB / 32, EMB / 32), 256, 0, stream>>>(Wk, WkT, EMB, EMB);
  transpose_cast_bf16<<<dim3(EMB / 32, EMB / 32), 256, 0, stream>>>(Wv, WvT, EMB, EMB);
  transpose_cast_bf16<<<dim3(EMB / 32, EMB / 32), 256, 0, stream>>>(Wo, WoT, EMB, EMB);
  transpose_cast_bf16<<<dim3(EMB / 32, FFDIM / 32), 256, 0, stream>>>(W1, W1T, EMB, FFDIM);
  transpose_cast_bf16<<<dim3(FFDIM / 32, EMB / 32), 256, 0, stream>>>(W2, W2T, FFDIM, EMB);

  // 2) LN1
  layernorm_bf16<<<ROWS, 256, 0, stream>>>(x, g1, be1, hbf);

  // 3) QKV projections
  const dim3 gN1(EMB / 128, ROWS / 128);
  gemm_bf16_wmma<MODE_QK><<<gN1, 256, 0, stream>>>(hbf, WqT, bq, nullptr, Qh, ROWS, EMB, EMB);
  gemm_bf16_wmma<MODE_QK><<<gN1, 256, 0, stream>>>(hbf, WkT, bk, nullptr, Kh, ROWS, EMB, EMB);
  gemm_bf16_wmma<MODE_VT><<<gN1, 256, 0, stream>>>(hbf, WvT, bv, nullptr, Vt, ROWS, EMB, EMB);

  // 4) causal flash attention
  attn_flash_wmma<<<dim3(SEQ / 64, BATCH * HEADS), 128, 0, stream>>>(Qh, Kh, Vt, ctx);

  // 5) output projection + residual (f32)
  gemm_bf16_wmma<MODE_F32RES><<<gN1, 256, 0, stream>>>(ctx, WoT, bo, x, x1, ROWS, EMB, EMB);

  // 6) LN2
  layernorm_bf16<<<ROWS, 256, 0, stream>>>(x1, g2, be2, hbf);

  // 7) FF1 + GELU
  gemm_bf16_wmma<MODE_GELU><<<dim3(FFDIM / 128, ROWS / 128), 256, 0, stream>>>(
      hbf, W1T, b1, nullptr, a1, ROWS, FFDIM, EMB);

  // 8) FF2 + residual -> out
  gemm_bf16_wmma<MODE_F32RES><<<gN1, 256, 0, stream>>>(a1, W2T, b2, x1, out, ROWS, EMB, FFDIM);
}
